// Moe_29214367548131
// MI455X (gfx1250) — matrix operand order repaired and checked by reading, not compiled, not run on hardware
//
#include <hip/hip_runtime.h>

// ---------------- problem dimensions ----------------
#define Bn   16      // batch
#define Mm   16      // tokens per sample
#define Dd   512     // model dim
#define Ne   16      // experts
#define Hh   682     // slots / hidden (ragged)
#define Hp   688     // slot axis padded to 43*16 (N-dim of GEMMs)
#define HP2  704     // hidden axis padded to 22*32 (K-dim of GEMM2)
#define OUTn 512
#define KKo  (Mm * Dd)   // 8192, K of output projection
#define NDk  (Ne * Hp)   // 11008, K of combine GEMM (344*32)

typedef __bf16 bfx;
typedef __attribute__((ext_vector_type(16))) __bf16 v16bf;
typedef __attribute__((ext_vector_type(8)))  float  v8f;

__device__ __forceinline__ bfx f2bf(float f) {
    union { float f; unsigned u; } v; v.f = f;
    unsigned r = (v.u + 0x7FFFu + ((v.u >> 16) & 1u)) >> 16;
    union { unsigned short s; bfx b; } o; o.s = (unsigned short)r;
    return o.b;
}
__device__ __forceinline__ v8f zero8() {
    v8f z;
#pragma unroll
    for (int i = 0; i < 8; ++i) z[i] = 0.f;
    return z;
}
__device__ __forceinline__ v8f wmma_bf16(v16bf a, v16bf b, v8f c) {
    return __builtin_amdgcn_wmma_f32_16x16x32_bf16(false, a, false, b, (short)0, c,
                                                   false, false);
}
// A fragment (contiguous k order; loadB uses the same): lane holds row (lane&15); element i maps to
// K = i + (hi?16:0)  -> two contiguous 8-elem (16B) chunks.
__device__ __forceinline__ v16bf loadA(const bfx* base, int hi) {
    union { uint4 q[2]; v16bf v; } u;
    u.q[0] = *(const uint4*)(base + hi * 16);
    u.q[1] = *(const uint4*)(base + hi * 16 + 8);
    return u.v;
}
// B fragment (contiguous k order, matching loadA): lane holds col (lane&15); element i maps to
// K = i + (hi?16:0) -> one contiguous 16-elem (32B) chunk.
__device__ __forceinline__ v16bf loadB(const bfx* base, int hi) {
    const uint4* p = (const uint4*)(base + hi * 16);
    union { uint4 q[2]; v16bf v; } u;
    u.q[0] = p[0];
    u.q[1] = p[1];
    return u.v;
}

// ================= prep kernels: pad + transpose + fp32->bf16 =================
__global__ void k_prep_x(const float* __restrict__ x, bfx* __restrict__ xbf,
                         bfx* __restrict__ xTp) {
    const int m = threadIdx.x;          // 0..31
    const int d = blockIdx.y;
    const int b = blockIdx.x;
    const float v = (m < Mm) ? x[((size_t)b * Mm + m) * Dd + d] : 0.f;
    xTp[((size_t)b * Dd + d) * 32 + m] = f2bf(v);
    if (m < Mm) xbf[((size_t)b * Mm + m) * Dd + d] = f2bf(v);
}
__global__ void k_prep_phi(const float* __restrict__ phi, bfx* __restrict__ phiT) {
    const int d = blockIdx.x * 256 + threadIdx.x;
    const int p = blockIdx.y, n = blockIdx.z;
    const float v = (p < Hh) ? phi[((size_t)d * Ne + n) * Hh + p] : 0.f;
    phiT[((size_t)n * Hp + p) * Dd + d] = f2bf(v);
}
__global__ void k_prep_w1(const float* __restrict__ W1, bfx* __restrict__ W1T) {
    const int d = blockIdx.x * 256 + threadIdx.x;
    const int h = blockIdx.y, n = blockIdx.z;
    const float v = (h < Hh) ? W1[((size_t)n * Dd + d) * Hh + h] : 0.f;
    W1T[((size_t)n * HP2 + h) * Dd + d] = f2bf(v);
}
__global__ void k_prep_w2(const float* __restrict__ W2, bfx* __restrict__ W2T) {
    const int h = blockIdx.x * 256 + threadIdx.x;
    if (h >= HP2) return;
    const int d = blockIdx.y, n = blockIdx.z;
    const float v = (h < Hh) ? W2[((size_t)n * Hh + h) * Dd + d] : 0.f;
    W2T[((size_t)n * Dd + d) * HP2 + h] = f2bf(v);
}
__global__ void k_prep_wout(const float* __restrict__ Wout, bfx* __restrict__ WoutT) {
    const int j = blockIdx.x * 256 + threadIdx.x;
    const int o = blockIdx.y;
    WoutT[(size_t)o * KKo + j] = f2bf(Wout[(size_t)j * OUTn + o]);
}

// ============ K1: logits[b][n][m][p] = sum_d x[b,m,d]*phi[d,n,p] ============
__global__ void k_logits(const bfx* __restrict__ xbf, const bfx* __restrict__ phiT,
                         float* __restrict__ logits) {
    const int lane = threadIdx.x & 31;
    const int r = lane & 15, hi = lane >> 4;
    const int p0 = blockIdx.x * 16;
    const int n = blockIdx.y, b = blockIdx.z;
    const bfx* aRow = xbf + ((size_t)b * Mm + r) * Dd;
    const bfx* bCol = phiT + ((size_t)n * Hp + p0 + r) * Dd;
    v8f acc = zero8();
    for (int k0 = 0; k0 < Dd; k0 += 32)
        acc = wmma_bf16(loadA(aRow + k0, hi), loadB(bCol + k0, hi), acc);
#pragma unroll
    for (int v = 0; v < 8; ++v) {
        const int m = v + (hi ? 8 : 0);
        logits[(((size_t)b * Ne + n) * Mm + m) * Hp + p0 + r] = acc[v];
    }
}

// ============ K2a: dispatch softmax over m -> DwT[b][n][p][m32] (bf16) ============
__global__ void k_dw(const float* __restrict__ logits, bfx* __restrict__ DwT) {
    const int p = blockIdx.x * blockDim.x + threadIdx.x;
    if (p >= Hp) return;
    const int n = blockIdx.y, b = blockIdx.z;
    const size_t base = ((size_t)b * Ne + n) * Mm * Hp + p;
    float mx = -1e30f;
#pragma unroll
    for (int m = 0; m < Mm; ++m) mx = fmaxf(mx, logits[base + (size_t)m * Hp]);
    float s = 0.f;
#pragma unroll
    for (int m = 0; m < Mm; ++m) s += __expf(logits[base + (size_t)m * Hp] - mx);
    const float inv = 1.f / s;
    const size_t ob = (((size_t)b * Ne + n) * Hp + p) * 32;
#pragma unroll
    for (int m = 0; m < Mm; ++m)
        DwT[ob + m] = f2bf(__expf(logits[base + (size_t)m * Hp] - mx) * inv);
#pragma unroll
    for (int m = Mm; m < 32; ++m) DwT[ob + m] = f2bf(0.f);
}

// ============ K2b: softmax over n -> s1 fp32 [B][M][N][Hp] ============
__global__ void k_cw1(const float* __restrict__ logits, float* __restrict__ s1) {
    const int p = blockIdx.x * blockDim.x + threadIdx.x;
    if (p >= Hp) return;
    const int m = blockIdx.y, b = blockIdx.z;
    float mx = -1e30f;
#pragma unroll
    for (int n = 0; n < Ne; ++n)
        mx = fmaxf(mx, logits[(((size_t)b * Ne + n) * Mm + m) * Hp + p]);
    float s = 0.f;
#pragma unroll
    for (int n = 0; n < Ne; ++n)
        s += __expf(logits[(((size_t)b * Ne + n) * Mm + m) * Hp + p] - mx);
    const float inv = 1.f / s;
#pragma unroll
    for (int n = 0; n < Ne; ++n)
        s1[(((size_t)b * Mm + m) * Ne + n) * Hp + p] =
            __expf(logits[(((size_t)b * Ne + n) * Mm + m) * Hp + p] - mx) * inv;
}

// ============ K2c: softmax over p -> CwB bf16 [B][M][NDk], padding zeroed ============
__global__ void k_cw2(const float* __restrict__ s1, bfx* __restrict__ CwB) {
    __shared__ float red[256];
    const int tid = threadIdx.x;
    const size_t base = (size_t)blockIdx.x * Hp;   // one (b,m,n) row
    float mx = -1e30f;
    for (int p = tid; p < Hh; p += 256) mx = fmaxf(mx, s1[base + p]);
    red[tid] = mx; __syncthreads();
    for (int s = 128; s > 0; s >>= 1) {
        if (tid < s) red[tid] = fmaxf(red[tid], red[tid + s]);
        __syncthreads();
    }
    mx = red[0]; __syncthreads();
    float sum = 0.f;
    for (int p = tid; p < Hh; p += 256) sum += __expf(s1[base + p] - mx);
    red[tid] = sum; __syncthreads();
    for (int s = 128; s > 0; s >>= 1) {
        if (tid < s) red[tid] += red[tid + s];
        __syncthreads();
    }
    const float inv = 1.f / red[0]; __syncthreads();
    for (int p = tid; p < Hp; p += 256) {
        const float v = (p < Hh) ? __expf(s1[base + p] - mx) * inv : 0.f;
        CwB[base + p] = f2bf(v);    // [B][M][N][Hp] flat == [B][M][NDk]
    }
}

// ============ K3: Xs[b,n,p,d] = sum_m DwT[b,n,p,m]*x[b,m,d]  (K=32, padded) ============
__global__ void k_xs(const bfx* __restrict__ xTp, const bfx* __restrict__ DwT,
                     bfx* __restrict__ Xs) {
    const int lane = threadIdx.x & 31;
    const int r = lane & 15, hi = lane >> 4;
    const int d0 = blockIdx.x * 16;
    const int p0 = blockIdx.y * 16;
    const int bn = blockIdx.z;
    const int b = bn / Ne, n = bn % Ne;
    const bfx* aRow = DwT + (((size_t)b * Ne + n) * Hp + p0 + r) * 32;
    const bfx* bCol = xTp + ((size_t)b * Dd + d0 + r) * 32;
    v8f acc = wmma_bf16(loadA(aRow, hi), loadB(bCol, hi), zero8());
#pragma unroll
    for (int v = 0; v < 8; ++v) {
        const int row = v + (hi ? 8 : 0);
        Xs[(((size_t)b * Ne + n) * Hp + p0 + row) * Dd + d0 + r] = f2bf(acc[v]);
    }
}

// ============ K4: fused expert MLP; hmid tile (16x704 bf16) in LDS ============
// Each wave owns a 16x64 output tile: A fragment loaded once, reused by 4 WMMAs.
__global__ void __launch_bounds__(128)
k_mlp(const bfx* __restrict__ Xs, const bfx* __restrict__ W1T, const float* __restrict__ b1,
      const bfx* __restrict__ W2T, const float* __restrict__ b2, bfx* __restrict__ YsT) {
    __shared__ bfx hs[16 * HP2];         // 22.5 KB of 320 KB WGP LDS
    const int lane = threadIdx.x & 31;
    const int wave = threadIdx.x >> 5;
    const int r = lane & 15, hi = lane >> 4;
    const int p0 = blockIdx.x * 16;
    const int n = blockIdx.y, b = blockIdx.z;
    const bfx* aRow = Xs + (((size_t)b * Ne + n) * Hp + p0 + r) * Dd;

    // phase 1: hmid[16, HP2] = relu(Xs_tile[16,512] @ W1_n + b1); 11 groups of 4 tiles
    for (int g = wave; g < HP2 / 64; g += 4) {
        const int hbase = g * 64;
        const bfx* bCol[4];
        v8f acc[4];
#pragma unroll
        for (int s = 0; s < 4; ++s) {
            bCol[s] = W1T + ((size_t)n * HP2 + hbase + s * 16 + r) * Dd;
            acc[s] = zero8();
        }
        for (int k0 = 0; k0 < Dd; k0 += 32) {
            const v16bf a = loadA(aRow + k0, hi);
#pragma unroll
            for (int s = 0; s < 4; ++s)
                acc[s] = wmma_bf16(a, loadB(bCol[s] + k0, hi), acc[s]);
        }
#pragma unroll
        for (int s = 0; s < 4; ++s) {
            const int h = hbase + s * 16 + r;
            const int hc = (h < Hh) ? h : (Hh - 1);
            const float bias = (h < Hh) ? b1[n * Hh + hc] : 0.f;
#pragma unroll
            for (int v = 0; v < 8; ++v) {
                const int row = v + (hi ? 8 : 0);
                hs[row * HP2 + h] = f2bf(fmaxf(acc[s][v] + bias, 0.f));
            }
        }
    }
    __syncthreads();
    // phase 2: Ys_tile[16,512] = hmid @ W2_n + b2 (transposed store); 8 groups of 4
    const bfx* aL = hs + r * HP2;
    for (int g = wave; g < Dd / 64; g += 4) {
        const int dbase = g * 64;
        const bfx* bCol[4];
        v8f acc[4];
#pragma unroll
        for (int s = 0; s < 4; ++s) {
            bCol[s] = W2T + ((size_t)n * Dd + dbase + s * 16 + r) * HP2;
            acc[s] = zero8();
        }
        for (int k0 = 0; k0 < HP2; k0 += 32) {
            const v16bf a = loadA(aL + k0, hi);
#pragma unroll
            for (int s = 0; s < 4; ++s)
                acc[s] = wmma_bf16(a, loadB(bCol[s] + k0, hi), acc[s]);
        }
#pragma unroll
        for (int s = 0; s < 4; ++s) {
            const int d = dbase + s * 16 + r;
            const float bias = b2[n * Dd + d];
            const size_t ob = ((size_t)b * Dd + d) * NDk + (size_t)n * Hp + p0;
#pragma unroll
            for (int v = 0; v < 8; ++v) {
                const int row = v + (hi ? 8 : 0);
                YsT[ob + row] = f2bf(acc[s][v] + bias);   // contiguous over v
            }
        }
    }
}

// ============ K5: Y[b,m,d] = sum_k Cw[b,m,k]*YsT[b,d,k], K = 11008 ============
__global__ void k_combine(const bfx* __restrict__ CwB, const bfx* __restrict__ YsT,
                          bfx* __restrict__ Yb) {
    const int lane = threadIdx.x & 31;
    const int r = lane & 15, hi = lane >> 4;
    const int dbase = blockIdx.x * 64;
    const int b = blockIdx.y;
    const bfx* aRow = CwB + ((size_t)b * Mm + r) * NDk;
    const bfx* bCol[4];
    v8f acc[4];
#pragma unroll
    for (int s = 0; s < 4; ++s) {
        bCol[s] = YsT + ((size_t)b * Dd + dbase + s * 16 + r) * NDk;
        acc[s] = zero8();
    }
    for (int k0 = 0; k0 < NDk; k0 += 32) {
        if (k0 + 256 < NDk) __builtin_prefetch(bCol[0] + k0 + 256, 0, 1);
        const v16bf a = loadA(aRow + k0, hi);
#pragma unroll
        for (int s = 0; s < 4; ++s)
            acc[s] = wmma_bf16(a, loadB(bCol[s] + k0, hi), acc[s]);
    }
#pragma unroll
    for (int s = 0; s < 4; ++s)
#pragma unroll
        for (int v = 0; v < 8; ++v) {
            const int row = v + (hi ? 8 : 0);
            Yb[(size_t)b * KKo + (size_t)row * Dd + dbase + s * 16 + r] = f2bf(acc[s][v]);
        }
}

// ============ K6: out[16,512] = Yb[16,8192] @ Wout + bout (fp32 out) ============
__global__ void k_out(const bfx* __restrict__ Yb, const bfx* __restrict__ WoutT,
                      const float* __restrict__ bout, float* __restrict__ out) {
    const int lane = threadIdx.x & 31;
    const int r = lane & 15, hi = lane >> 4;
    const int dbase = blockIdx.x * 64;
    const bfx* aRow = Yb + (size_t)r * KKo;
    const bfx* bCol[4];
    v8f acc[4];
#pragma unroll
    for (int s = 0; s < 4; ++s) {
        bCol[s] = WoutT + (size_t)(dbase + s * 16 + r) * KKo;
        acc[s] = zero8();
    }
    for (int k0 = 0; k0 < KKo; k0 += 32) {
        const v16bf a = loadA(aRow + k0, hi);
#pragma unroll
        for (int s = 0; s < 4; ++s)
            acc[s] = wmma_bf16(a, loadB(bCol[s] + k0, hi), acc[s]);
    }
#pragma unroll
    for (int s = 0; s < 4; ++s) {
        const float bias = bout[dbase + s * 16 + r];
#pragma unroll
        for (int v = 0; v < 8; ++v) {
            const int row = v + (hi ? 8 : 0);
            out[(size_t)row * OUTn + dbase + s * 16 + r] = acc[s][v] + bias;
        }
    }
}

// ---------------- workspace layout (bytes, all 16B aligned) ----------------
static constexpr size_t E_LOG  = (size_t)Bn * Ne * Mm * Hp;      // fp32
static constexpr size_t E_S1   = E_LOG;                          // fp32
static constexpr size_t E_CWB  = (size_t)Bn * Mm * NDk;          // bf16
static constexpr size_t E_DWT  = (size_t)Bn * Ne * Hp * 32;      // bf16
static constexpr size_t E_XBF  = (size_t)Bn * Mm * Dd;           // bf16
static constexpr size_t E_XTP  = (size_t)Bn * Dd * 32;           // bf16
static constexpr size_t E_PHIT = (size_t)Ne * Hp * Dd;           // bf16
static constexpr size_t E_W1T  = (size_t)Ne * HP2 * Dd;          // bf16
static constexpr size_t E_W2T  = (size_t)Ne * Dd * HP2;          // bf16
static constexpr size_t E_WOT  = (size_t)OUTn * KKo;             // bf16
static constexpr size_t E_XS   = (size_t)Bn * Ne * Hp * Dd;      // bf16
static constexpr size_t E_YST  = (size_t)Bn * Dd * NDk;          // bf16
static constexpr size_t E_YB   = (size_t)Bn * KKo;               // bf16

static constexpr size_t O_LOG  = 0;
static constexpr size_t O_S1   = O_LOG  + E_LOG * 4;
static constexpr size_t O_CWB  = O_S1   + E_S1 * 4;
static constexpr size_t O_DWT  = O_CWB  + E_CWB * 2;
static constexpr size_t O_XBF  = O_DWT  + E_DWT * 2;
static constexpr size_t O_XTP  = O_XBF  + E_XBF * 2;
static constexpr size_t O_PHIT = O_XTP  + E_XTP * 2;
static constexpr size_t O_W1T  = O_PHIT + E_PHIT * 2;
static constexpr size_t O_W2T  = O_W1T  + E_W1T * 2;
static constexpr size_t O_WOT  = O_W2T  + E_W2T * 2;
static constexpr size_t O_XS   = O_WOT  + E_WOT * 2;
static constexpr size_t O_YST  = O_XS   + E_XS * 2;
static constexpr size_t O_YB   = O_YST  + E_YST * 2;

extern "C" void kernel_launch(void* const* d_in, const int* in_sizes, int n_in,
                              void* d_out, int out_size, void* d_ws, size_t ws_size,
                              hipStream_t stream) {
    const float* x    = (const float*)d_in[0];
    const float* phi  = (const float*)d_in[1];
    const float* W1   = (const float*)d_in[2];
    const float* b1   = (const float*)d_in[3];
    const float* W2   = (const float*)d_in[4];
    const float* b2   = (const float*)d_in[5];
    const float* Wout = (const float*)d_in[6];
    const float* bout = (const float*)d_in[7];
    float* out = (float*)d_out;

    char* ws = (char*)d_ws;
    float* logits = (float*)(ws + O_LOG);
    float* s1     = (float*)(ws + O_S1);
    bfx* CwB  = (bfx*)(ws + O_CWB);
    bfx* DwT  = (bfx*)(ws + O_DWT);
    bfx* xbf  = (bfx*)(ws + O_XBF);
    bfx* xTp  = (bfx*)(ws + O_XTP);
    bfx* phiT = (bfx*)(ws + O_PHIT);
    bfx* W1T  = (bfx*)(ws + O_W1T);
    bfx* W2T  = (bfx*)(ws + O_W2T);
    bfx* WoT  = (bfx*)(ws + O_WOT);
    bfx* Xs   = (bfx*)(ws + O_XS);
    bfx* YsT  = (bfx*)(ws + O_YST);
    bfx* Yb   = (bfx*)(ws + O_YB);

    // prep: pad + transpose + cvt to bf16 (removes all guards/cvt from GEMM loops)
    k_prep_x   <<<dim3(Bn, Dd),            32,  0, stream>>>(x, xbf, xTp);
    k_prep_phi <<<dim3(Dd / 256, Hp, Ne),  256, 0, stream>>>(phi, phiT);
    k_prep_w1  <<<dim3(Dd / 256, HP2, Ne), 256, 0, stream>>>(W1, W1T);
    k_prep_w2  <<<dim3((HP2 + 255) / 256, Dd, Ne), 256, 0, stream>>>(W2, W2T);
    k_prep_wout<<<dim3(KKo / 256, OUTn),   256, 0, stream>>>(Wout, WoT);

    k_logits <<<dim3(Hp / 16, Ne, Bn), 32,  0, stream>>>(xbf, phiT, logits);
    k_dw     <<<dim3((Hp + 127) / 128, Ne, Bn), 128, 0, stream>>>(logits, DwT);
    k_cw1    <<<dim3((Hp + 127) / 128, Mm, Bn), 128, 0, stream>>>(logits, s1);
    k_cw2    <<<dim3(Bn * Mm * Ne), 256, 0, stream>>>(s1, CwB);
    k_xs     <<<dim3(Dd / 16, Hp / 16, Bn * Ne), 32, 0, stream>>>(xTp, DwT, Xs);
    k_mlp    <<<dim3(Hp / 16, Ne, Bn), 128, 0, stream>>>(Xs, W1T, b1, W2T, b2, YsT);
    k_combine<<<dim3(Dd / 64, Bn), 32, 0, stream>>>(CwB, YsT, Yb);
    k_out    <<<dim3(OUTn / 64), 32, 0, stream>>>(Yb, WoT, bout, out);
}
